// MultiHeadAttentionBlock_429496729863
// MI455X (gfx1250) — matrix operand order repaired and checked by reading, not compiled, not run on hardware
//
#include <hip/hip_runtime.h>
#include <stdint.h>

#define BATCH 2
#define SEQ   2048
#define DMODEL 1024
#define NHEAD 16
#define DHEAD 64

typedef __attribute__((ext_vector_type(16))) _Float16     v16h;
typedef __attribute__((ext_vector_type(8)))  _Float16     v8h;
typedef __attribute__((ext_vector_type(8)))  float        v8f;
typedef __attribute__((ext_vector_type(4)))  unsigned int v4u;
typedef __attribute__((ext_vector_type(8)))  int          v8i;
typedef __attribute__((ext_vector_type(4)))  int          v4i;

#if __has_builtin(__builtin_amdgcn_tensor_load_to_lds)
#define HAS_TDM 1
#else
#define HAS_TDM 0
#endif

// ---------------------------------------------------------------------------
// Fragment loaders (wave32 WMMA 16x16x32 f16 layouts, CDNA5 ISA 7.12.2)
// A (16x32, MxK): lane = 16c+m.  a[j] = A[m][ j<8 ? 8c+j : 16+8c+(j-8) ]
// B (32x16, KxN): lane = 16c+n.  b[j] = B[ j<8 ? 8c+j : 16+8c+(j-8) ][n]
// C/D (16x16):    lane = 16*hi+t, VGPR r -> C[r+8*hi][t]
// ---------------------------------------------------------------------------
static __device__ inline v16h frag_join(v8h lo, v8h hi) {
    v16h r;
#pragma unroll
    for (int i = 0; i < 8; ++i) { r[i] = lo[i]; r[i + 8] = hi[i]; }
    return r;
}
static __device__ inline v16h load_frag_a(const _Float16* rowptr, int c) {
    v8h lo = *(const v8h*)(rowptr + 8 * c);
    v8h hi = *(const v8h*)(rowptr + 16 + 8 * c);
    return frag_join(lo, hi);
}
static __device__ inline v16h load_frag_b(const _Float16* rowptr, int c) {
    v8h lo = *(const v8h*)(rowptr + 8 * c);
    v8h hi = *(const v8h*)(rowptr + 16 + 8 * c);
    return frag_join(lo, hi);
}

// ---------------------------------------------------------------------------
// TDM: 2-D tensor_load_to_lds per CDNA5 D# layout (ISA 08 §8.3/8.4).
// tileD0 elems contiguous (k), tileD1 rows; data_size=2B; LDS rows padded
// 64B data + 16B pad -> stride 40 halves. Tracked by TENSORcnt.
// ---------------------------------------------------------------------------
#if HAS_TDM
static __device__ inline void tdm_load_2d(unsigned int ldsOff, const _Float16* gptr,
                                          unsigned int tileD0, unsigned int tileD1,
                                          unsigned int tensorD0, unsigned int tensorD1,
                                          unsigned int strideElems)
{
    uint64_t ga = (uint64_t)(uintptr_t)gptr;
    v4u g0;
    g0[0] = 1u;                                            // count=1, no gather
    g0[1] = ldsOff;                                        // lds_addr (bytes)
    g0[2] = (unsigned int)(ga & 0xFFFFFFFFu);              // global_addr[31:0]
    g0[3] = (unsigned int)((ga >> 32) & 0x1FFFFFFu)        // global_addr[56:32]
            | (2u << 30);                                  // type = 2 ("image")
    v8i g1;
    g1[0] = (int)((1u << 16)        // data_size = 2 bytes
                | (1u << 20)        // pad_enable
                | (3u << 22)        // pad_interval: 64B
                | (3u << 25));      // pad_amount: 4 DWORDs = 16B
    g1[1] = (int)((tensorD0 & 0xFFFFu) << 16);
    g1[2] = (int)(((tensorD0 >> 16) & 0xFFFFu) | ((tensorD1 & 0xFFFFu) << 16));
    g1[3] = (int)(((tensorD1 >> 16) & 0xFFFFu) | ((tileD0 & 0xFFFFu) << 16));
    g1[4] = (int)(tileD1 & 0xFFFFu);                       // tile_dim1, tile_dim2=0
    g1[5] = (int)strideElems;                              // tensor_dim0_stride[31:0]
    g1[6] = 0;
    g1[7] = 0;
    v4i z4 = {0, 0, 0, 0};
#if defined(__clang_major__) && __clang_major__ >= 23
    v8i z8 = {0, 0, 0, 0, 0, 0, 0, 0};
    __builtin_amdgcn_tensor_load_to_lds(g0, g1, z4, z4, z8, 0);
#else
    __builtin_amdgcn_tensor_load_to_lds(g0, g1, z4, z4, 0);
#endif
}
#endif

// ---------------------------------------------------------------------------
// Precision pre-passes: f32 -> f16 planes, and f32 W[K][N] -> f16 Wt[N][K].
// ---------------------------------------------------------------------------
__global__ __launch_bounds__(256) void cvt_f32_f16(const float* __restrict__ s,
                                                   _Float16* __restrict__ d, int n)
{
    const int idx = (blockIdx.x * 256 + threadIdx.x) * 8;
    if (idx < n) {
#pragma unroll
        for (int j = 0; j < 8; ++j) d[idx + j] = (_Float16)s[idx + j];
    }
}

__global__ __launch_bounds__(256) void transpose_cvt_f16(
    const float* __restrict__ W, _Float16* __restrict__ Wt, int Kd, int Nd)
{
    __shared__ _Float16 tileS[32][33];
    const int k0 = blockIdx.y * 32, n0 = blockIdx.x * 32;
    const int r = threadIdx.x >> 3, cb = (threadIdx.x & 7) * 4;
#pragma unroll
    for (int i = 0; i < 4; ++i)
        tileS[r][cb + i] = (_Float16)W[(size_t)(k0 + r) * Nd + n0 + cb + i];
    __syncthreads();
#pragma unroll
    for (int i = 0; i < 4; ++i)
        Wt[(size_t)(n0 + r) * Kd + k0 + cb + i] = tileS[cb + i][r];
}

// ---------------------------------------------------------------------------
// WMMA GEMM: C[M,N] = A[M,K] @ Wt[N,K]^T, A & Wt in f16 (k-contiguous rows).
// 256 threads = 8 waves; block tile 128x64, BK=32; wave w owns rows 16w..16w+15.
// TDM double-buffers both LDS tiles; wave 0 issues + waits TENSORcnt.
// VTRS: scatter-store C transposed per head: VT[(b*H+h)*64+d][s].
// ---------------------------------------------------------------------------
template <typename OutT, bool VTRS>
__global__ __launch_bounds__(256) void gemm_wmma_tdm(
    const _Float16* __restrict__ A, const _Float16* __restrict__ Wt,
    OutT* __restrict__ C, int M, int N, int K)
{
    const int BKP = 40;                          // padded LDS stride (halves)
    __shared__ _Float16 As2[2][128 * BKP];
    __shared__ _Float16 Bt2[2][64 * BKP];

    const int tid  = threadIdx.x;
    const int lane = tid & 31, wave = tid >> 5;
    const int m0 = blockIdx.y * 128, n0 = blockIdx.x * 64;
    const int t = lane & 15, c = lane >> 4;

    const int iters = K / 32;
    v8f acc[4] = {};

#if HAS_TDM
    if (wave == 0) {
        tdm_load_2d((unsigned int)(uintptr_t)&As2[0][0], A + (size_t)m0 * K,
                    32, 128, (unsigned int)K, (unsigned int)M, (unsigned int)K);
        tdm_load_2d((unsigned int)(uintptr_t)&Bt2[0][0], Wt + (size_t)n0 * K,
                    32, 64, (unsigned int)K, (unsigned int)N, (unsigned int)K);
    }
#endif

    for (int it = 0; it < iters; ++it) {
        const int p = it & 1;
#if HAS_TDM
        if (wave == 0) {
            if (it + 1 < iters) {
                const int kt1 = (it + 1) * 32;
                tdm_load_2d((unsigned int)(uintptr_t)&As2[p ^ 1][0],
                            A + (size_t)m0 * K + kt1,
                            32, 128, (unsigned int)K, (unsigned int)M, (unsigned int)K);
                tdm_load_2d((unsigned int)(uintptr_t)&Bt2[p ^ 1][0],
                            Wt + (size_t)n0 * K + kt1,
                            32, 64, (unsigned int)K, (unsigned int)N, (unsigned int)K);
                __builtin_amdgcn_s_wait_tensorcnt(2);   // current tiles complete
            } else {
                __builtin_amdgcn_s_wait_tensorcnt(0);
            }
        }
#else
        {   // manual staging fallback (no TDM builtin on this toolchain)
            const int kt = it * 32;
            const int row = tid >> 1, colb = (tid & 1) * 16;
            const _Float16* srcA = A + (size_t)(m0 + row) * K + kt + colb;
            _Float16* dstA = &As2[p][row * BKP + colb];
#pragma unroll
            for (int i = 0; i < 16; ++i) dstA[i] = srcA[i];
            const int wrow = tid >> 2, wcb = (tid & 3) * 8;
            const _Float16* srcW = Wt + (size_t)(n0 + wrow) * K + kt + wcb;
            _Float16* dstW = &Bt2[p][wrow * BKP + wcb];
#pragma unroll
            for (int i = 0; i < 8; ++i) dstW[i] = srcW[i];
        }
#endif
        __syncthreads();                 // tiles for parity p are valid

        v16h afrag = load_frag_a(&As2[p][(wave * 16 + t) * BKP], c);
#pragma unroll
        for (int nbk = 0; nbk < 4; ++nbk) {
            v16h bfrag = load_frag_b(&Bt2[p][(nbk * 16 + t) * BKP], c);
            acc[nbk] = __builtin_amdgcn_wmma_f32_16x16x32_f16(
                false, afrag, false, bfrag, (short)0, acc[nbk], false, false);
        }
        __syncthreads();                 // done reading before p^1 reuse
    }

#pragma unroll
    for (int nbk = 0; nbk < 4; ++nbk)
#pragma unroll
        for (int r = 0; r < 8; ++r) {
            const int row = m0 + wave * 16 + r + 8 * c;   // = b*SEQ + s
            const int col = n0 + nbk * 16 + t;            // = h*64 + d
            if (VTRS) {
                const int bb = row >> 11, ss = row & (SEQ - 1);   // SEQ = 2048
                const int hh = col >> 6,  dd = col & (DHEAD - 1);
                C[(((size_t)bb * NHEAD + hh) * DHEAD + dd) * SEQ + ss] =
                    (OutT)acc[nbk][r];
            } else {
                C[(size_t)row * N + col] = (OutT)acc[nbk][r];
            }
        }
}

// ---------------------------------------------------------------------------
// Flash attention: Q/K planes [B*S, DMODEL] f16, V transposed [B*H*64, SEQ]
// f16. One wave per 16-row q-tile of one head, 32 keys per step, online
// softmax with f32 accumulators. V fragments load straight from VT (two
// b128 global loads per lane) -- no LDS staging for V.
// ---------------------------------------------------------------------------
__global__ __launch_bounds__(128) void flash_attn_wmma(
    const _Float16* __restrict__ Qp, const _Float16* __restrict__ Kp,
    const _Float16* __restrict__ VT, const int* __restrict__ mask,
    _Float16* __restrict__ X)
{
    __shared__ _Float16 Pb[4][16 * 32];          // per-wave P re-layout buffer

    const int tid  = threadIdx.x;
    const int lane = tid & 31, wave = tid >> 5;
    const int tileIdx = blockIdx.x * 4 + wave;
    const int tile = tileIdx & (SEQ / 16 - 1);
    const int bh   = tileIdx / (SEQ / 16);
    const int h = bh % NHEAD, b = bh / NHEAD;
    const int q0 = tile * 16;
    const int t = lane & 15, c = lane >> 4;

    const size_t rs = DMODEL;

    const _Float16* Qbase = Qp + ((size_t)(b * SEQ + q0 + t)) * rs + h * DHEAD;
    v16h qf[2];
#pragma unroll
    for (int kc = 0; kc < 2; ++kc) qf[kc] = load_frag_a(Qbase + kc * 32, c);

    const _Float16* Vbase = VT + ((size_t)(b * NHEAD + h)) * DHEAD * SEQ;

    float runmax[8], runsum[8];
    v8f o[4] = {};
#pragma unroll
    for (int r = 0; r < 8; ++r) { runmax[r] = -3.0e38f; runsum[r] = 0.0f; }

    const int* mrow = mask + ((size_t)b * SEQ + q0) * SEQ;

    for (int key0 = 0; key0 < SEQ; key0 += 32) {
        v8f s[2] = {};
#pragma unroll
        for (int half = 0; half < 2; ++half) {
            const _Float16* Kbase =
                Kp + ((size_t)(b * SEQ + key0 + half * 16 + t)) * rs + h * DHEAD;
#pragma unroll
            for (int kc = 0; kc < 2; ++kc) {
                v16h kf = load_frag_b(Kbase + kc * 32, c);
                s[half] = __builtin_amdgcn_wmma_f32_16x16x32_f16(
                    false, qf[kc], false, kf, (short)0, s[half], false, false);
            }
        }

        float p0[8], p1[8], alpha[8], tmax[8];
#pragma unroll
        for (int r = 0; r < 8; ++r) {
            const int qrow = r + 8 * c;
            const int mk0 = mrow[(size_t)qrow * SEQ + key0 + t];
            const int mk1 = mrow[(size_t)qrow * SEQ + key0 + 16 + t];
            const float s0 = mk0 ? s[0][r] * 0.125f : -1.0e9f;
            const float s1 = mk1 ? s[1][r] * 0.125f : -1.0e9f;
            s[0][r] = s0; s[1][r] = s1;
            tmax[r] = fmaxf(s0, s1);
        }
#pragma unroll
        for (int r = 0; r < 8; ++r)
#pragma unroll
            for (int off = 1; off < 16; off <<= 1)      // stays in 16-lane half
                tmax[r] = fmaxf(tmax[r], __shfl_xor(tmax[r], off, 32));
#pragma unroll
        for (int r = 0; r < 8; ++r) {
            const float nm = fmaxf(runmax[r], tmax[r]);
            alpha[r] = __expf(runmax[r] - nm);
            runmax[r] = nm;
            p0[r] = __expf(s[0][r] - nm);
            p1[r] = __expf(s[1][r] - nm);
            float ps = p0[r] + p1[r];
#pragma unroll
            for (int off = 1; off < 16; off <<= 1)
                ps += __shfl_xor(ps, off, 32);
            runsum[r] = runsum[r] * alpha[r] + ps;
        }

#pragma unroll
        for (int r = 0; r < 8; ++r) {
            const int qrow = r + 8 * c;
            Pb[wave][qrow * 32 + t]      = (_Float16)p0[r];
            Pb[wave][qrow * 32 + 16 + t] = (_Float16)p1[r];
        }
        v16h pf = load_frag_a(&Pb[wave][t * 32], c);     // same-wave LDS RAW: in-order

#pragma unroll
        for (int nbk = 0; nbk < 4; ++nbk) {
#pragma unroll
            for (int r = 0; r < 8; ++r) o[nbk][r] *= alpha[r];
            // B fragment straight from VT: lane row d = nbk*16+t, keys contiguous
            v16h vf = load_frag_b(Vbase + (size_t)(nbk * 16 + t) * SEQ + key0, c);
            o[nbk] = __builtin_amdgcn_wmma_f32_16x16x32_f16(
                false, pf, false, vf, (short)0, o[nbk], false, false);
        }
    }

    _Float16* Xbase = X + ((size_t)(b * SEQ + q0)) * DMODEL + h * DHEAD;
#pragma unroll
    for (int nbk = 0; nbk < 4; ++nbk)
#pragma unroll
        for (int r = 0; r < 8; ++r) {
            const int qrow = r + 8 * c;
            Xbase[(size_t)qrow * DMODEL + nbk * 16 + t] =
                (_Float16)(o[nbk][r] / runsum[r]);
        }
}

// ---------------------------------------------------------------------------
extern "C" void kernel_launch(void* const* d_in, const int* in_sizes, int n_in,
                              void* d_out, int out_size, void* d_ws, size_t ws_size,
                              hipStream_t stream) {
    (void)in_sizes; (void)n_in; (void)out_size; (void)ws_size;
    const float* q    = (const float*)d_in[0];
    const float* k    = (const float*)d_in[1];
    const float* v    = (const float*)d_in[2];
    const int*   mask = (const int*)d_in[3];
    const float* Wq   = (const float*)d_in[4];
    const float* Wk   = (const float*)d_in[5];
    const float* Wv   = (const float*)d_in[6];
    const float* Wo   = (const float*)d_in[7];
    float* out = (float*)d_out;

    const int M = BATCH * SEQ, N = DMODEL, K = DMODEL;
    const size_t plane  = (size_t)M * N;        // 4 Mi f16 elems
    const size_t wplane = (size_t)K * N;        // 1 Mi f16 elems

    _Float16* Qw  = (_Float16*)d_ws;            // attention inputs / outputs
    _Float16* Kw  = Qw + plane;
    _Float16* VTw = Kw + plane;                 // V, head-transposed [B*H*64][SEQ]
    _Float16* Xw  = VTw + plane;
    _Float16* qh  = Xw + plane;                 // f16 copies of activations
    _Float16* kh  = qh + plane;
    _Float16* vh  = kh + plane;
    _Float16* WqT = vh + plane;                 // f16 transposed weights [N][K]
    _Float16* WkT = WqT + wplane;
    _Float16* WvT = WkT + wplane;
    _Float16* WoT = WvT + wplane;               // total ws: 64 MB

    // --- precision pre-passes ------------------------------------------------
    const int nplane = (int)plane;
    dim3 cblk(256), cgrid(nplane / (256 * 8));
    cvt_f32_f16<<<cgrid, cblk, 0, stream>>>(q, qh, nplane);
    cvt_f32_f16<<<cgrid, cblk, 0, stream>>>(k, kh, nplane);
    cvt_f32_f16<<<cgrid, cblk, 0, stream>>>(v, vh, nplane);

    dim3 tgrid(N / 32, K / 32);
    transpose_cvt_f16<<<tgrid, cblk, 0, stream>>>(Wq, WqT, K, N);
    transpose_cvt_f16<<<tgrid, cblk, 0, stream>>>(Wk, WkT, K, N);
    transpose_cvt_f16<<<tgrid, cblk, 0, stream>>>(Wv, WvT, K, N);
    transpose_cvt_f16<<<tgrid, cblk, 0, stream>>>(Wo, WoT, K, N);

    // --- projections (TDM-staged WMMA GEMMs) --------------------------------
    dim3 gblk(256), ggrid(N / 64, M / 128);
    gemm_wmma_tdm<_Float16, false><<<ggrid, gblk, 0, stream>>>(qh, WqT, Qw, M, N, K);
    gemm_wmma_tdm<_Float16, false><<<ggrid, gblk, 0, stream>>>(kh, WkT, Kw, M, N, K);
    gemm_wmma_tdm<_Float16, true ><<<ggrid, gblk, 0, stream>>>(vh, WvT, VTw, M, N, K);

    // --- attention -----------------------------------------------------------
    const int qtiles = BATCH * NHEAD * (SEQ / 16);
    flash_attn_wmma<<<qtiles / 4, 128, 0, stream>>>(Qw, Kw, VTw, mask, Xw);

    // --- output projection ---------------------------------------------------
    gemm_wmma_tdm<float, false><<<ggrid, gblk, 0, stream>>>(Xw, WoT, out, M, N, K);
}
